// CompressiveLayer_1116691496982
// MI455X (gfx1250) — matrix operand order repaired and checked by reading, not compiled, not run on hardware
//
#include <hip/hip_runtime.h>
#include <cstdint>

// ---------------------------------------------------------------------------
// CDNA5 (gfx1250) bf16 WMMA types & helpers
// ---------------------------------------------------------------------------
typedef __attribute__((ext_vector_type(16))) __bf16 bf16x16;
typedef __attribute__((ext_vector_type(8)))  __bf16 bf16x8;
typedef __attribute__((ext_vector_type(8)))  float  floatx8;

union FragU { bf16x16 v; bf16x8 h[2]; };

__device__ __forceinline__ floatx8 wmma_bf16(bf16x16 a, bf16x16 b, floatx8 c) {
  // D = A(16x32 bf16) x B(32x16 bf16) + C(16x16 f32)
  return __builtin_amdgcn_wmma_f32_16x16x32_bf16(false, a, false, b,
                                                 (short)0, c, false, false);
}

// A-operand layout (16-bit A 16x32): lane group g: VGPR0-3 = K[8g..8g+7],
// VGPR4-7 = K[16+8g..16+8g+7].  `tile` points at subtile row 0, K-contiguous.
__device__ __forceinline__ bf16x16 ld_afrag(const __bf16* tile, int ld, int k0, int lane) {
  const int g = lane >> 4, m = lane & 15;
  const __bf16* p = tile + m * ld + k0 + g * 8;
  FragU f;
  f.h[0] = *(const bf16x8*)(p);
  f.h[1] = *(const bf16x8*)(p + 16);
  return f.v;
}

// B-operand layout (16-bit B 32x16): lane group g: VGPR0-3 = K[8g..8g+7],
// VGPR4-7 = K[16+8g..16+8g+7].  `tile` = subtile col 0, K-contiguous rows.
__device__ __forceinline__ bf16x16 ld_bfrag(const __bf16* tile, int ld, int k0, int lane) {
  const int g = lane >> 4, n = lane & 15;
  const __bf16* p = tile + n * ld + k0 + g * 8;
  FragU f;
  f.h[0] = *(const bf16x8*)(p);
  f.h[1] = *(const bf16x8*)(p + 16);
  return f.v;
}

__device__ __forceinline__ void cvt4(__bf16* p, float4 f) {
  p[0] = (__bf16)f.x; p[1] = (__bf16)f.y; p[2] = (__bf16)f.z; p[3] = (__bf16)f.w;
}

// ---------------------------------------------------------------------------
// Async global->LDS copies (GLOBAL_LOAD_ASYNC_TO_LDS_B128, ASYNCcnt-tracked).
// Probe result: builtin exists, param 1 = v4i in address_space(1).
// ---------------------------------------------------------------------------
#if defined(__gfx1250__) && __has_builtin(__builtin_amdgcn_global_load_async_to_lds_b128)
#define HAS_ASYNC_LDS 1
#else
#define HAS_ASYNC_LDS 0
#endif

typedef int v4i_vs __attribute__((__vector_size__(4 * sizeof(int))));
typedef __attribute__((address_space(1))) v4i_vs as1_v4i;
typedef __attribute__((address_space(3))) v4i_vs as3_v4i;

__device__ __forceinline__ void async_cp16(const __bf16* g, __bf16* l) {
#if HAS_ASYNC_LDS
  __builtin_amdgcn_global_load_async_to_lds_b128(
      (as1_v4i*)(unsigned long long)(uintptr_t)g,
      (as3_v4i*)(unsigned)(uintptr_t)l, 0, 0);
#else
  *(bf16x8*)l = *(const bf16x8*)g;
#endif
}

__device__ __forceinline__ void async_wait0() {
#if HAS_ASYNC_LDS
#if __has_builtin(__builtin_amdgcn_s_wait_asynccnt)
  __builtin_amdgcn_s_wait_asynccnt(0);
#else
  asm volatile("s_wait_asynccnt 0x0" ::: "memory");
#endif
#endif
}

__device__ __forceinline__ void async_wait4() {
#if HAS_ASYNC_LDS
#if __has_builtin(__builtin_amdgcn_s_wait_asynccnt)
  __builtin_amdgcn_s_wait_asynccnt(4);
#else
  asm volatile("s_wait_asynccnt 0x4" ::: "memory");
#endif
#endif
}

// ---------------------------------------------------------------------------
// Problem constants
// ---------------------------------------------------------------------------
#define QLEN_C  1024
#define KLEN_C  2560
#define MTOT_C  1536   // KLEN - QLEN
#define BSZ_C   2
#define DM_C    1024
#define NH_C    16
#define DH_C    64

// ---------------------------------------------------------------------------
// One-shot conversion kernels (f32 -> bf16), streaming / bandwidth-bound.
// ---------------------------------------------------------------------------
__global__ __launch_bounds__(256)
void cast_kernel(const float* __restrict__ in, __bf16* __restrict__ out, int n)
{
  int i = (blockIdx.x * 256 + threadIdx.x) * 4;
  if (i < n) cvt4(out + i, *(const float4*)(in + i));
}

// in: K x N (f32, row-major) -> out: N x K (bf16).  64x64 tiles via LDS.
__global__ __launch_bounds__(256)
void transpose_cast_kernel(const float* __restrict__ in, __bf16* __restrict__ out,
                           int K, int N)
{
  __shared__ float tile[64][65];
  const int k0 = blockIdx.y * 64, n0 = blockIdx.x * 64;
  const int tr = threadIdx.x >> 6, tc = threadIdx.x & 63;
#pragma unroll
  for (int p = 0; p < 16; ++p)
    tile[tr + p * 4][tc] = in[(size_t)(k0 + tr + p * 4) * N + n0 + tc];
  __syncthreads();
#pragma unroll
  for (int p = 0; p < 16; ++p)
    out[(size_t)(n0 + tr + p * 4) * K + k0 + tc] = (__bf16)tile[tc][tr + p * 4];
}

// ---------------------------------------------------------------------------
// bf16 WMMA GEMM, all-bf16 operands, async double-buffered tiles.
//   A:  M x K bf16 (row-major, K contiguous)
//   Bt: N x K bf16 (pre-transposed weight, K contiguous)
//   EP: 0 = store, 1 = +res(f32), 2 = +bias & relu, 3 = +bias +res
//   OUTBF: 1 = store bf16, 0 = store f32
// Block: 256 thr (8 waves 4x2), tile 128x128, K-step 32, 2 LDS buffers.
// ---------------------------------------------------------------------------
#define GTM 128
#define GTK 32
#define GAP 40   // padded LDS row stride (elements), multiple of 8

template<int EP, int OUTBF>
__global__ __launch_bounds__(256)
void gemm_kernel(const __bf16* __restrict__ A, const __bf16* __restrict__ Bt,
                 void* __restrict__ Cout, int M, int N, int K,
                 const float* __restrict__ res, const float* __restrict__ bias)
{
  __shared__ __bf16 As[2][GTM * GAP];
  __shared__ __bf16 Bs[2][GTM * GAP];

  const int t    = threadIdx.x;
  const int lane = t & 31;
  const int wv   = t >> 5;
  const int wr   = wv >> 1;        // wave row 0..3 (M)
  const int wc   = wv & 1;         // wave col 0..1 (N)
  const int bm   = blockIdx.y * GTM;
  const int bn   = blockIdx.x * GTM;

  // tile-fill mapping: thread t covers row (t>>1) of BOTH tiles, 16 elements
  const int tr = t >> 1;
  const int kh = (t & 1) * 16;
  const __bf16* ga = A  + (size_t)(bm + tr) * K + kh;
  const __bf16* gb = Bt + (size_t)(bn + tr) * K + kh;

  floatx8 acc[2][4];
#pragma unroll
  for (int i = 0; i < 2; ++i)
#pragma unroll
    for (int j = 0; j < 4; ++j) acc[i][j] = {};

  auto issue = [&](int k0, int buf) {
    async_cp16(ga + k0,     &As[buf][tr * GAP + kh]);
    async_cp16(ga + k0 + 8, &As[buf][tr * GAP + kh + 8]);
    async_cp16(gb + k0,     &Bs[buf][tr * GAP + kh]);
    async_cp16(gb + k0 + 8, &Bs[buf][tr * GAP + kh + 8]);
  };

  issue(0, 0);
  const int nk = K / GTK;
  for (int i = 0; i < nk; ++i) {
    const int cur = i & 1;
    if (i + 1 < nk) { issue((i + 1) * GTK, cur ^ 1); async_wait4(); }
    else            { async_wait0(); }
    __syncthreads();                 // all waves' tile copies resident

    bf16x16 a0 = ld_afrag(&As[cur][(wr * 32     ) * GAP], GAP, 0, lane);
    bf16x16 a1 = ld_afrag(&As[cur][(wr * 32 + 16) * GAP], GAP, 0, lane);
#pragma unroll
    for (int ni = 0; ni < 4; ++ni) {
      bf16x16 bfr = ld_bfrag(&Bs[cur][(wc * 64 + ni * 16) * GAP], GAP, 0, lane);
      acc[0][ni] = wmma_bf16(a0, bfr, acc[0][ni]);
      acc[1][ni] = wmma_bf16(a1, bfr, acc[1][ni]);
    }
    __syncthreads();                 // done reading buf before it is re-filled
  }

  // Epilogue: C/D layout -> row m = r + 8*(lane>>4), col n = lane&15
  const int g = lane >> 4, nn = lane & 15;
#pragma unroll
  for (int mi = 0; mi < 2; ++mi)
#pragma unroll
    for (int ni = 0; ni < 4; ++ni)
#pragma unroll
      for (int r = 0; r < 8; ++r) {
        int m = bm + wr * 32 + mi * 16 + r + 8 * g;
        int n = bn + wc * 64 + ni * 16 + nn;
        float v = acc[mi][ni][r];
        if (EP == 1) v += res[(size_t)m * N + n];
        if (EP == 2) { v += bias[n]; v = v > 0.f ? v : 0.f; }
        if (EP == 3) v += bias[n] + res[(size_t)m * N + n];
        if (OUTBF) ((__bf16*)Cout)[(size_t)m * N + n] = (__bf16)v;
        else       ((float*)Cout)[(size_t)m * N + n] = v;
      }
}

// ---------------------------------------------------------------------------
// Fused flash attention with Transformer-XL relative shift (all-bf16 inputs).
//   score[i,j] = ((q_i+r_w_bias)·k_j + (q_i+r_r_bias)·r_k[j+QLEN-1-i]) / 8
//   rel-shift handled as a 64x128 WMMA matmul G + LDS gather G[v, u-v+63].
// Grid: (QLEN/64, NH, BSZ), block 128 (4 waves; wave w owns q rows 16w..16w+15)
// ---------------------------------------------------------------------------
__global__ __launch_bounds__(128)
void attn_kernel(const __bf16* __restrict__ whb, const __bf16* __restrict__ rkb,
                 const float* __restrict__ rwb, const float* __restrict__ rrb,
                 __bf16* __restrict__ avb)
{
  __shared__ __bf16 qw[64 * 72], qr[64 * 72];
  __shared__ __bf16 Ks[64 * 72], Vs[64 * 72], Ps[64 * 72];
  __shared__ __bf16 Rs[128 * 72];
  __shared__ float  Gs[64 * 132];

  const int i0 = blockIdx.x * 64;
  const int nh = blockIdx.y;
  const int b  = blockIdx.z;
  const int t = threadIdx.x, lane = t & 31, wv = t >> 5;
  const int g = lane >> 4, ln = lane & 15;

  // ---- q block: bf16 -> f32, add both bias variants, back to bf16 ----
  {
    int row = t >> 1, ch = (t & 1) * 32;
    size_t base = ((size_t)(1536 + i0 + row) * BSZ_C + b) * (3 * DM_C) + nh * DH_C + ch;
#pragma unroll
    for (int e = 0; e < 32; e += 8) {
      bf16x8 f = *(const bf16x8*)(whb + base + e);
#pragma unroll
      for (int x = 0; x < 8; ++x) {
        float qv = (float)f[x];
        qw[row * 72 + ch + e + x] = (__bf16)(qv + rwb[nh * DH_C + ch + e + x]);
        qr[row * 72 + ch + e + x] = (__bf16)(qv + rrb[nh * DH_C + ch + e + x]);
      }
    }
  }

  float mrow[8], lrow[8];
  floatx8 O[4];
#pragma unroll
  for (int r = 0; r < 8; ++r) { mrow[r] = -3.0e38f; lrow[r] = 0.f; }
#pragma unroll
  for (int d = 0; d < 4; ++d) O[d] = {};

  const float scale = 0.125f;              // 1/sqrt(64)
  const int   c0    = QLEN_C - 1 - i0;     // shift const: idx = j0 + c0 + u - vb
  const int   jend  = i0 + 64 + MTOT_C;    // multiple of 64, <= KLEN

  for (int j0 = 0; j0 < jend; j0 += 64) {
    __syncthreads();  // previous tile fully consumed before overwrite

    const int jr = t >> 1, ch = (t & 1) * 32;
    // ---- K tile: straight bf16 rows -> async copy to LDS ----
    {
      const __bf16* gk = whb + ((size_t)(j0 + jr) * BSZ_C + b) * (3 * DM_C)
                       + DM_C + nh * DH_C + ch;
      __bf16* lk = &Ks[jr * 72 + ch];
      async_cp16(gk,      lk);
      async_cp16(gk + 8,  lk + 8);
      async_cp16(gk + 16, lk + 16);
      async_cp16(gk + 24, lk + 24);
    }
    // ---- Rs: 128-row shifted r_k slice, clamped (OOB rows are masked) ----
#pragma unroll
    for (int pass = 0; pass < 2; ++pass) {
      int rr  = jr + pass * 64;
      int src = j0 + c0 - 63 + rr;
      src = src < 0 ? 0 : (src > KLEN_C - 1 ? KLEN_C - 1 : src);
      const __bf16* gr = rkb + (size_t)src * DM_C + nh * DH_C + ch;
      __bf16* lr = &Rs[rr * 72 + ch];
      async_cp16(gr,      lr);
      async_cp16(gr + 8,  lr + 8);
      async_cp16(gr + 16, lr + 16);
      async_cp16(gr + 24, lr + 24);
    }
    // ---- V tile: transposed in LDS (Vs[d][j]) -> VALU scatter ----
    {
      const __bf16* gv = whb + ((size_t)(j0 + jr) * BSZ_C + b) * (3 * DM_C)
                       + 2 * DM_C + nh * DH_C + ch;
#pragma unroll
      for (int e = 0; e < 32; e += 8) {
        bf16x8 f = *(const bf16x8*)(gv + e);
#pragma unroll
        for (int x = 0; x < 8; ++x) Vs[(ch + e + x) * 72 + jr] = f[x];
      }
    }
    async_wait0();
    __syncthreads();

    bf16x16 aw0 = ld_afrag(&qw[16 * wv * 72], 72, 0,  lane);
    bf16x16 aw1 = ld_afrag(&qw[16 * wv * 72], 72, 32, lane);
    bf16x16 ar0 = ld_afrag(&qr[16 * wv * 72], 72, 0,  lane);
    bf16x16 ar1 = ld_afrag(&qr[16 * wv * 72], 72, 32, lane);

    // AC = (q + r_w_bias) @ K^T  (16x64 per wave)
    floatx8 ac[4];
#pragma unroll
    for (int jn = 0; jn < 4; ++jn) {
      floatx8 s = {};
      s = wmma_bf16(aw0, ld_bfrag(&Ks[jn * 16 * 72], 72, 0,  lane), s);
      s = wmma_bf16(aw1, ld_bfrag(&Ks[jn * 16 * 72], 72, 32, lane), s);
      ac[jn] = s;
    }
    // G = (q + r_r_bias) @ Rslice^T  (16x128 per wave) -> wave-private Gs rows
#pragma unroll
    for (int wn = 0; wn < 8; ++wn) {
      floatx8 gg = {};
      gg = wmma_bf16(ar0, ld_bfrag(&Rs[wn * 16 * 72], 72, 0,  lane), gg);
      gg = wmma_bf16(ar1, ld_bfrag(&Rs[wn * 16 * 72], 72, 32, lane), gg);
#pragma unroll
      for (int r = 0; r < 8; ++r)
        Gs[(16 * wv + r + 8 * g) * 132 + wn * 16 + ln] = gg[r];
    }
    // S = (AC + rel-shifted BD) * scale, causal mask
#pragma unroll
    for (int jn = 0; jn < 4; ++jn)
#pragma unroll
      for (int r = 0; r < 8; ++r) {
        int vb = 16 * wv + r + 8 * g;          // block-local q row
        int u  = jn * 16 + ln;                 // block-local key col
        float bd = Gs[vb * 132 + (u - vb + 63)];
        float s  = (ac[jn][r] + bd) * scale;
        if (j0 + u > i0 + vb + MTOT_C) s = -1.0e30f;
        ac[jn][r] = s;
      }
    // ---- online softmax (each 16-lane group holds one row) ----
#pragma unroll
    for (int r = 0; r < 8; ++r) {
      float mx = ac[0][r];
      mx = fmaxf(mx, ac[1][r]); mx = fmaxf(mx, ac[2][r]); mx = fmaxf(mx, ac[3][r]);
#pragma unroll
      for (int dd = 1; dd < 16; dd <<= 1) mx = fmaxf(mx, __shfl_xor(mx, dd, 32));
      float mnew = fmaxf(mrow[r], mx);
      float corr = __expf(mrow[r] - mnew);
      mrow[r] = mnew;
      float rs = 0.f;
#pragma unroll
      for (int jn = 0; jn < 4; ++jn) {
        float p = __expf(ac[jn][r] - mnew);
        ac[jn][r] = p;
        rs += p;
      }
#pragma unroll
      for (int dd = 1; dd < 16; dd <<= 1) rs += __shfl_xor(rs, dd, 32);
      lrow[r] = lrow[r] * corr + rs;
#pragma unroll
      for (int dn = 0; dn < 4; ++dn) O[dn][r] *= corr;
    }
    // ---- stage P (bf16) through wave-private LDS rows, then O += P @ V ----
#pragma unroll
    for (int jn = 0; jn < 4; ++jn)
#pragma unroll
      for (int r = 0; r < 8; ++r)
        Ps[(16 * wv + r + 8 * g) * 72 + jn * 16 + ln] = (__bf16)ac[jn][r];

    bf16x16 pa0 = ld_afrag(&Ps[16 * wv * 72], 72, 0,  lane);
    bf16x16 pa1 = ld_afrag(&Ps[16 * wv * 72], 72, 32, lane);
#pragma unroll
    for (int dn = 0; dn < 4; ++dn) {
      O[dn] = wmma_bf16(pa0, ld_bfrag(&Vs[dn * 16 * 72], 72, 0,  lane), O[dn]);
      O[dn] = wmma_bf16(pa1, ld_bfrag(&Vs[dn * 16 * 72], 72, 32, lane), O[dn]);
    }
  }

  // ---- normalize and write attn_vec (qlen, bsz, nh*dh) in bf16 ----
#pragma unroll
  for (int dn = 0; dn < 4; ++dn)
#pragma unroll
    for (int r = 0; r < 8; ++r) {
      int vb = 16 * wv + r + 8 * g;
      float val = O[dn][r] / lrow[r];
      avb[((size_t)(i0 + vb) * BSZ_C + b) * DM_C + nh * DH_C + dn * 16 + ln] =
          (__bf16)val;
    }
}

// ---------------------------------------------------------------------------
// Row LayerNorm over D=1024.  Optionally also emits a bf16 copy (GEMM operand).
// ---------------------------------------------------------------------------
template<int WB>
__global__ __launch_bounds__(256)
void ln_kernel(const float* __restrict__ x, const float* __restrict__ sc,
               const float* __restrict__ bi, float* __restrict__ y,
               __bf16* __restrict__ yb)
{
  const float* xr = x + (size_t)blockIdx.x * DM_C;
  float*       yr = y + (size_t)blockIdx.x * DM_C;
  __bf16*      br = WB ? yb + (size_t)blockIdx.x * DM_C : nullptr;
  const int t = threadIdx.x, lane = t & 31, wv = t >> 5;
  __shared__ float red[8];

  float v0 = xr[t], v1 = xr[t + 256], v2 = xr[t + 512], v3 = xr[t + 768];
  float s = v0 + v1 + v2 + v3;
#pragma unroll
  for (int d = 1; d < 32; d <<= 1) s += __shfl_xor(s, d, 32);
  if (lane == 0) red[wv] = s;
  __syncthreads();
  float tot = 0.f;
#pragma unroll
  for (int i = 0; i < 8; ++i) tot += red[i];
  const float mu = tot * (1.f / (float)DM_C);

  float d0 = v0 - mu, d1 = v1 - mu, d2 = v2 - mu, d3 = v3 - mu;
  float q = d0 * d0 + d1 * d1 + d2 * d2 + d3 * d3;
#pragma unroll
  for (int d = 1; d < 32; d <<= 1) q += __shfl_xor(q, d, 32);
  __syncthreads();
  if (lane == 0) red[wv] = q;
  __syncthreads();
  float qt = 0.f;
#pragma unroll
  for (int i = 0; i < 8; ++i) qt += red[i];
  const float rstd = rsqrtf(qt * (1.f / (float)DM_C) + 1e-5f);

#pragma unroll
  for (int p = 0; p < 4; ++p) {
    float dv = (p == 0 ? d0 : p == 1 ? d1 : p == 2 ? d2 : d3);
    int idx = t + p * 256;
    float o = dv * rstd * sc[idx] + bi[idx];
    yr[idx] = o;
    if (WB) br[idx] = (__bf16)o;
  }
}

// ---------------------------------------------------------------------------
// Host-side orchestration
// ---------------------------------------------------------------------------
extern "C" void kernel_launch(void* const* d_in, const int* in_sizes, int n_in,
                              void* d_out, int out_size, void* d_ws, size_t ws_size,
                              hipStream_t stream)
{
  (void)in_sizes; (void)n_in; (void)out_size; (void)ws_size;

  const float* input_ids = (const float*)d_in[0];   // (1024, 2, 1024)
  const float* pos_emb   = (const float*)d_in[1];   // (2560, 1024)
  const float* mem       = (const float*)d_in[2];   // (1024, 2, 1024)
  const float* c_mem     = (const float*)d_in[3];   // (512, 2, 1024)
  /* d_in[4] attn_mask: recomputed analytically in-kernel */
  const float* qkv_w     = (const float*)d_in[5];   // (1024, 3072)
  const float* r_w       = (const float*)d_in[6];   // (1024, 1024)
  const float* o_w       = (const float*)d_in[7];   // (1024, 1024)
  const float* r_w_bias  = (const float*)d_in[8];   // (16, 64)
  const float* r_r_bias  = (const float*)d_in[9];   // (16, 64)
  const float* ln_a_sc   = (const float*)d_in[10];
  const float* ln_a_bi   = (const float*)d_in[11];
  const float* ff_w1     = (const float*)d_in[12];  // (1024, 4096)
  const float* ff_b1     = (const float*)d_in[13];
  const float* ff_w2     = (const float*)d_in[14];  // (4096, 1024)
  const float* ff_b2     = (const float*)d_in[15];
  const float* ln_f_sc   = (const float*)d_in[16];
  const float* ln_f_bi   = (const float*)d_in[17];

  // ---- workspace layout (256B aligned chunks) ----
  char* wp = (char*)d_ws;
  auto alloc = [&](size_t bytes) {
    char* p = wp; wp += (bytes + 255) & ~(size_t)255; return p;
  };
  __bf16* catb  = (__bf16*)alloc((size_t)5120 * 1024 * 2);  // concat A, bf16
  __bf16* posb  = (__bf16*)alloc((size_t)2560 * 1024 * 2);
  __bf16* qkvT  = (__bf16*)alloc((size_t)3072 * 1024 * 2);  // weights, N x K
  __bf16* rwT   = (__bf16*)alloc((size_t)1024 * 1024 * 2);
  __bf16* owT   = (__bf16*)alloc((size_t)1024 * 1024 * 2);
  __bf16* fw1T  = (__bf16*)alloc((size_t)4096 * 1024 * 2);
  __bf16* fw2T  = (__bf16*)alloc((size_t)1024 * 4096 * 2);
  __bf16* whb   = (__bf16*)alloc((size_t)5120 * 3072 * 2);  // w_heads bf16
  __bf16* rkb   = (__bf16*)alloc((size_t)2560 * 1024 * 2);
  __bf16* avb   = (__bf16*)alloc((size_t)2048 * 1024 * 2);
  float*  x1    = (float*) alloc((size_t)2048 * 1024 * 4);  // reused as x2
  float*  ares  = (float*) alloc((size_t)2048 * 1024 * 4);
  __bf16* aresb = (__bf16*)alloc((size_t)2048 * 1024 * 2);
  __bf16* hb    = (__bf16*)alloc((size_t)2048 * 4096 * 2);
  float*  x2    = x1;

  // ---- 0) one-shot casts / weight transposes (bandwidth-trivial) ----
  cast_kernel<<<2048, 256, 0, stream>>>(mem,       catb,                2097152);
  cast_kernel<<<1024, 256, 0, stream>>>(c_mem,     catb + 2048 * 1024,  1048576);
  cast_kernel<<<2048, 256, 0, stream>>>(input_ids, catb + 3072 * 1024,  2097152);
  cast_kernel<<<2560, 256, 0, stream>>>(pos_emb,   posb,                2621440);
  transpose_cast_kernel<<<dim3(48, 16), 256, 0, stream>>>(qkv_w, qkvT, 1024, 3072);
  transpose_cast_kernel<<<dim3(16, 16), 256, 0, stream>>>(r_w,   rwT,  1024, 1024);
  transpose_cast_kernel<<<dim3(16, 16), 256, 0, stream>>>(o_w,   owT,  1024, 1024);
  transpose_cast_kernel<<<dim3(64, 16), 256, 0, stream>>>(ff_w1, fw1T, 1024, 4096);
  transpose_cast_kernel<<<dim3(16, 64), 256, 0, stream>>>(ff_w2, fw2T, 4096, 1024);

  // 1) w_heads = concat(mem,c_mem,input) @ qkv_w   (5120 x 3072, K=1024) -> bf16
  gemm_kernel<0, 1><<<dim3(24, 40), 256, 0, stream>>>(
      catb, qkvT, whb, 5120, 3072, 1024, nullptr, nullptr);

  // 2) r_k = pos_emb @ r_w                          (2560 x 1024, K=1024) -> bf16
  gemm_kernel<0, 1><<<dim3(8, 20), 256, 0, stream>>>(
      posb, rwT, rkb, 2560, 1024, 1024, nullptr, nullptr);

  // 3) fused rel-attention -> attn_vec bf16
  attn_kernel<<<dim3(QLEN_C / 64, NH_C, BSZ_C), 128, 0, stream>>>(
      whb, rkb, r_w_bias, r_r_bias, avb);

  // 4) x1 = input + attn_vec @ o_w                  (2048 x 1024, K=1024) -> f32
  gemm_kernel<1, 0><<<dim3(8, 16), 256, 0, stream>>>(
      avb, owT, x1, 2048, 1024, 1024, input_ids, nullptr);

  // 5) attn_res = LN(x1)  (f32 + bf16 copies)
  ln_kernel<1><<<2048, 256, 0, stream>>>(x1, ln_a_sc, ln_a_bi, ares, aresb);

  // 6) h = relu(attn_res @ ff_w1 + b1)              (2048 x 4096, K=1024) -> bf16
  gemm_kernel<2, 1><<<dim3(32, 16), 256, 0, stream>>>(
      aresb, fw1T, hb, 2048, 4096, 1024, nullptr, ff_b1);

  // 7) x2 = attn_res + h @ ff_w2 + b2               (2048 x 1024, K=4096) -> f32
  gemm_kernel<3, 0><<<dim3(8, 16), 256, 0, stream>>>(
      hb, fw2T, x2, 2048, 1024, 4096, ares, ff_b2);

  // 8) out = LN(x2)
  ln_kernel<0><<<2048, 256, 0, stream>>>(x2, ln_f_sc, ln_f_bi, (float*)d_out, nullptr);
}
